// GNN_Encoder_39539468927050
// MI455X (gfx1250) — matrix operand order repaired and checked by reading, not compiled, not run on hardware
//
#include <hip/hip_runtime.h>
#include <hip/hip_bf16.h>

// ---------------------------------------------------------------------------
// GNN encoder for MI455X (gfx1250).  GEMMs use v_wmma_f32_16x16x32_bf16.
// ---------------------------------------------------------------------------

typedef __attribute__((ext_vector_type(16))) __bf16 v16bf;
typedef __attribute__((ext_vector_type(2)))  __bf16 v2bf;
typedef __attribute__((ext_vector_type(8)))  float  v8f;
typedef __attribute__((ext_vector_type(4)))  unsigned int u32x4;

struct Frag32B { u32x4 lo; u32x4 hi; };   // 32 bytes = 16 bf16

__device__ __forceinline__ unsigned short f2bf(float f) {
  union { float f; unsigned u; } v; v.f = f;
  unsigned u = v.u;
  unsigned r = u + 0x7FFFu + ((u >> 16) & 1u);   // round-to-nearest-even
  return (unsigned short)(r >> 16);
}

// pack two f32 -> packed bf16 pair (hardware v_cvt_pk_bf16_f32 when available)
__device__ __forceinline__ unsigned pack_bf16x2(float x, float y) {
#if __has_builtin(__builtin_amdgcn_cvt_pk_bf16_f32)
  v2bf t = __builtin_amdgcn_cvt_pk_bf16_f32(x, y);
  return __builtin_bit_cast(unsigned, t);
#else
  return (unsigned)f2bf(x) | ((unsigned)f2bf(y) << 16);
#endif
}

// -------------------------------- small utility kernels --------------------

// W [K][N] f32  ->  Wt [N][K] bf16   (one-time, tiny)
__global__ void cvt_transpose_bf16(const float* __restrict__ W,
                                   unsigned short* __restrict__ Wt,
                                   int K, int N) {
  int i = blockIdx.x * blockDim.x + threadIdx.x;
  if (i >= K * N) return;
  int k = i / N, n = i % N;
  Wt[(size_t)n * K + k] = f2bf(W[i]);
}

// h[i,:] = emb1[x[sni[i],0],:] + emb2[x[sni[i],1],:]      (64 lanes / node, float4)
__global__ void embed_sub(const int* __restrict__ x,
                          const int* __restrict__ sni,
                          const float* __restrict__ emb1,
                          const float* __restrict__ emb2,
                          float* __restrict__ h, int nsub) {
  int g = blockIdx.x * blockDim.x + threadIdx.x;
  int i = g >> 6;
  if (i >= nsub) return;
  int j = (g & 63) << 2;
  int node = sni[i];
  int a = x[2 * node], c = x[2 * node + 1];
  float4 e1 = *(const float4*)(emb1 + (size_t)a * 256 + j);
  float4 e2 = *(const float4*)(emb2 + (size_t)c * 256 + j);
  float4 o; o.x = e1.x + e2.x; o.y = e1.y + e2.y; o.z = e1.z + e2.z; o.w = e1.w + e2.w;
  *(float4*)(h + (size_t)i * 256 + j) = o;
}

// out_cat[i, 0:256] = emb1[x[i,0],:] + emb2[x[i,1],:]   (row stride 512)
__global__ void embed_origin(const int* __restrict__ x,
                             const float* __restrict__ emb1,
                             const float* __restrict__ emb2,
                             float* __restrict__ out_cat, int n) {
  int g = blockIdx.x * blockDim.x + threadIdx.x;
  int i = g >> 6;
  if (i >= n) return;
  int j = (g & 63) << 2;
  int a = x[2 * i], c = x[2 * i + 1];
  float4 e1 = *(const float4*)(emb1 + (size_t)a * 256 + j);
  float4 e2 = *(const float4*)(emb2 + (size_t)c * 256 + j);
  float4 o; o.x = e1.x + e2.x; o.y = e1.y + e2.y; o.z = e1.z + e2.z; o.w = e1.w + e2.w;
  *(float4*)(out_cat + (size_t)i * 512 + j) = o;
}

// edge scatter:  agg[dst] += h[src] + ee1[a0] + ee2[a1]   (self loops appended)
__global__ void scatter_edges(const float* __restrict__ h,
                              const int* __restrict__ src,
                              const int* __restrict__ dst,
                              const int* __restrict__ eattr,
                              const float* __restrict__ ee1,   // [6,256]
                              const float* __restrict__ ee2,   // [3,256]
                              float* __restrict__ agg,
                              int esub, int etot) {
  int g = blockIdx.x * blockDim.x + threadIdx.x;
  int e = g >> 6;
  if (e >= etot) return;
  int j = (g & 63) << 2;
  int s, d, a0, a1;
  if (e < esub) {
    s = src[e]; d = dst[e];
    a0 = eattr[2 * e]; a1 = eattr[2 * e + 1];
  } else {                         // self loop: bond type 4, direction 0
    s = d = e - esub; a0 = 4; a1 = 0;
  }
  float4 hv = *(const float4*)(h + (size_t)s * 256 + j);
  float4 v1 = *(const float4*)(ee1 + (size_t)a0 * 256 + j);
  float4 v2 = *(const float4*)(ee2 + (size_t)a1 * 256 + j);
  float* p = agg + (size_t)d * 256 + j;
  atomicAdd(p + 0, hv.x + v1.x + v2.x);
  atomicAdd(p + 1, hv.y + v1.y + v2.y);
  atomicAdd(p + 2, hv.z + v1.z + v2.z);
  atomicAdd(p + 3, hv.w + v1.w + v2.w);
}

// pooling: out_cat[ind[i], 256: ] += h[i,:]
__global__ void pool_nodes(const float* __restrict__ h,
                           const int* __restrict__ ind,
                           float* __restrict__ out_cat, int nsub) {
  int g = blockIdx.x * blockDim.x + threadIdx.x;
  int i = g >> 6;
  if (i >= nsub) return;
  int j = (g & 63) << 2;
  int d = ind[i];
  float4 hv = *(const float4*)(h + (size_t)i * 256 + j);
  float* p = out_cat + (size_t)d * 512 + 256 + j;
  atomicAdd(p + 0, hv.x);
  atomicAdd(p + 1, hv.y);
  atomicAdd(p + 2, hv.z);
  atomicAdd(p + 3, hv.w);
}

// per-column sum / sumsq partials (coalesced rows), atomically combined
__global__ void col_stats(const float* __restrict__ X, int rows, int cols,
                          int rowsPerBlock,
                          float* __restrict__ sums, float* __restrict__ sumsq) {
  int t = threadIdx.x;
  int r0 = blockIdx.x * rowsPerBlock;
  int r1 = r0 + rowsPerBlock; if (r1 > rows) r1 = rows;
  for (int c = t; c < cols; c += blockDim.x) {
    float s = 0.f, q = 0.f;
    for (int r = r0; r < r1; ++r) {
      float v = X[(size_t)r * cols + c];
      s += v; q += v * v;
    }
    atomicAdd(&sums[c], s);
    atomicAdd(&sumsq[c], q);
  }
}

// scale[c] = g*rsqrt(var+eps); shift[c] = b - mu*scale
__global__ void bn_finalize(const float* __restrict__ sums,
                            const float* __restrict__ sumsq,
                            const float* __restrict__ gw,
                            const float* __restrict__ bw,
                            float rowsInv, int cols,
                            float* __restrict__ scale, float* __restrict__ shift) {
  int c = blockIdx.x * blockDim.x + threadIdx.x;
  if (c >= cols) return;
  float mu  = sums[c] * rowsInv;
  float var = sumsq[c] * rowsInv - mu * mu;
  float s = gw[c] * rsqrtf(var + 1e-5f);
  scale[c] = s;
  shift[c] = bw[c] - mu * s;
}

__global__ void bn_apply(float* __restrict__ X,
                         const float* __restrict__ scale,
                         const float* __restrict__ shift,
                         int rows, int cols) {
  int g = blockIdx.x * blockDim.x + threadIdx.x;
  if (g >= rows * cols) return;
  int c = g % cols;
  X[g] = X[g] * scale[c] + shift[c];
}

// -------------------------------- WMMA GEMM --------------------------------
// C[M,N] = act( A @ B + bias )
// A : f32 row-major [M,K], converted to bf16 while staging into LDS (float4 loads,
//     hardware v_cvt_pk_bf16_f32 pack)
// Bt: bf16 [N][K] (weights pre-transposed on device -> b128 loads + b128 DS stores)
// Block tile 64(M) x 256(N), K-step 32, 256 threads = 8 waves.
// Wave tile 32(M) x 64(N): 8 accumulators, 8x v_wmma per K-step,
// fed by 12 ds_load_b128 (2 A frags + 4 B frags).
#define BLK_M 64
#define BLK_N 256
#define BLK_K 32

__global__ __launch_bounds__(256)
void gemm_bf16_wmma(const float* __restrict__ A,
                    const unsigned short* __restrict__ Bt,   // [N][K] bf16
                    const float* __restrict__ bias,
                    float* __restrict__ C,
                    int M, int N, int K, int relu) {
  __shared__ alignas(32) unsigned short As[BLK_M][BLK_K];   //  4 KB
  __shared__ alignas(32) unsigned short Bs[BLK_N][BLK_K];   // 16 KB

  const int tid  = threadIdx.x;
  const int lane = tid & 31;
  const int wave = tid >> 5;
  const int wm   = wave & 1;        // 0..1 -> M sub-tile (32 rows each)
  const int wn   = wave >> 1;       // 0..3 -> 64-col strip
  const int r    = lane & 15;
  const int hi   = lane >> 4;

  const int mBase = blockIdx.x * BLK_M;
  const int nBase = blockIdx.y * BLK_N;

  v8f acc[2][4] = {};

  for (int kt = 0; kt < K; kt += BLK_K) {
    // ---- stage A tile: 2048 f32, float4 per thread x2, pack to bf16 ----
#pragma unroll
    for (int i = 0; i < 2; ++i) {
      int q   = i * 256 + tid;          // 16-byte chunk id
      int row = q >> 3;                 // 8 chunks per 32-wide row
      int c4  = (q & 7) << 2;
      int gr  = mBase + row;
      float4 v = make_float4(0.f, 0.f, 0.f, 0.f);
      if (gr < M) v = *(const float4*)(A + (size_t)gr * K + kt + c4);
      *(uint2*)&As[row][c4] = make_uint2(pack_bf16x2(v.x, v.y),
                                         pack_bf16x2(v.z, v.w));
    }
    // ---- stage B tile: 8192 bf16, b128 per thread x4, contiguous both ways ----
#pragma unroll
    for (int i = 0; i < 4; ++i) {
      int q   = i * 256 + tid;          // 16-byte chunk id (8 bf16)
      int n   = q >> 2;                 // 4 chunks per 32-wide row
      int c8  = (q & 3) << 3;
      u32x4 v = *(const u32x4*)(Bt + (size_t)(nBase + n) * K + kt + c8);
      *(u32x4*)&Bs[n][c8] = v;
    }
    // ---- warm L2/L0 for the next K tile (global_prefetch, no counter cost) ----
    if (kt + BLK_K < K) {
      int arow = mBase + (tid >> 3);
      if (arow < M) __builtin_prefetch(A + (size_t)arow * K + kt + BLK_K, 0, 1);
      __builtin_prefetch(Bt + (size_t)(nBase + (tid >> 2)) * K + kt + BLK_K, 0, 1);
    }
    __syncthreads();

    // ---- A fragments (contiguous K layout: lane half hi holds K 16*hi..16*hi+15) ----
    v16bf aF[2];
#pragma unroll
    for (int mi = 0; mi < 2; ++mi) {
      Frag32B fa;
      fa.lo = *(const u32x4*)&As[wm * 32 + mi * 16 + r][16 * hi];
      fa.hi = *(const u32x4*)&As[wm * 32 + mi * 16 + r][16 * hi + 8];
      aF[mi] = __builtin_bit_cast(v16bf, fa);
    }

#pragma unroll
    for (int a = 0; a < 4; ++a) {
      int c = wn * 64 + a * 16 + r;     // lane (mod 16) = B column
      v16bf bF = *(const v16bf*)&Bs[c][16 * hi];
#pragma unroll
      for (int mi = 0; mi < 2; ++mi) {
        acc[mi][a] = __builtin_amdgcn_wmma_f32_16x16x32_bf16(
            false, aF[mi], false, bF, (short)0, acc[mi][a], false, false);
      }
    }
    __syncthreads();
  }

  // ---- epilogue: C VGPR i -> row i + 8*hi, col = lane&15 ----
#pragma unroll
  for (int a = 0; a < 4; ++a) {
    int gc = nBase + wn * 64 + a * 16 + r;
    float bv = bias[gc];
#pragma unroll
    for (int mi = 0; mi < 2; ++mi) {
#pragma unroll
      for (int i = 0; i < 8; ++i) {
        int gr = mBase + wm * 32 + mi * 16 + hi * 8 + i;
        if (gr < M) {
          float v = acc[mi][a][i] + bv;
          if (relu) v = fmaxf(v, 0.f);
          C[(size_t)gr * N + gc] = v;
        }
      }
    }
  }
}

// -------------------------------- host launcher ----------------------------

extern "C" void kernel_launch(void* const* d_in, const int* in_sizes, int n_in,
                              void* d_out, int out_size, void* d_ws, size_t ws_size,
                              hipStream_t stream) {
  const int N = 20000, NSUB = 100000, ESUB = 300000, D = 256, L = 2;
  const int ETOT = ESUB + NSUB;
  (void)in_sizes; (void)n_in; (void)out_size; (void)ws_size; (void)D;

  const int*   x        = (const int*)d_in[0];
  const int*   sni      = (const int*)d_in[1];
  const int*   eidx     = (const int*)d_in[2];           // [2, ESUB]
  const int*   eattr    = (const int*)d_in[3];           // [ESUB, 2]
  const int*   ind      = (const int*)d_in[4];
  const float* emb1     = (const float*)d_in[5];
  const float* emb2     = (const float*)d_in[6];
  const float* edge_e1  = (const float*)d_in[7];         // [L,6,256]
  const float* edge_e2  = (const float*)d_in[8];         // [L,3,256]
  const float* W1       = (const float*)d_in[9];         // [L,256,512]
  const float* b1       = (const float*)d_in[10];        // [L,512]
  const float* W2       = (const float*)d_in[11];        // [L,512,256]
  const float* b2       = (const float*)d_in[12];        // [L,256]
  const float* bn_cat_g = (const float*)d_in[13];
  const float* bn_cat_b = (const float*)d_in[14];
  const float* Wp       = (const float*)d_in[15];        // [512,256]
  const float* bp       = (const float*)d_in[16];
  const float* norm_g   = (const float*)d_in[17];
  const float* norm_b   = (const float*)d_in[18];

  const int* src = eidx;
  const int* dst = eidx + ESUB;

  // ---- carve workspace ----
  char* p = (char*)d_ws;
  auto alloc = [&](size_t bytes) {
    void* q = (void*)p;
    p += (bytes + 255) & ~(size_t)255;
    return q;
  };
  float* h       = (float*)alloc((size_t)NSUB * 256 * 4);   // 100 MB
  float* agg     = (float*)alloc((size_t)NSUB * 256 * 4);   // 100 MB
  float* g1      = (float*)alloc((size_t)NSUB * 512 * 4);   // 200 MB
  float* out_cat = (float*)alloc((size_t)N * 512 * 4);      //  41 MB
  unsigned short* wB1t = (unsigned short*)alloc((size_t)L * 256 * 512 * 2); // [512][256] per layer
  unsigned short* wB2t = (unsigned short*)alloc((size_t)L * 512 * 256 * 2); // [256][512] per layer
  unsigned short* wpBt = (unsigned short*)alloc((size_t)512 * 256 * 2);     // [256][512]
  float* bn1buf = (float*)alloc(1024 * 4);   // sums1[512] | sumsq1[512]
  float* scale1 = (float*)alloc(512 * 4);
  float* shift1 = (float*)alloc(512 * 4);
  float* bn2buf = (float*)alloc(512 * 4);    // sums2[256] | sumsq2[256]
  float* scale2 = (float*)alloc(256 * 4);
  float* shift2 = (float*)alloc(256 * 4);
  float* sums1 = bn1buf, *sumsq1 = bn1buf + 512;
  float* sums2 = bn2buf, *sumsq2 = bn2buf + 256;

  // ---- pre-transpose weights to bf16 [N][K] ----
  {
    int nEl = 256 * 512;
    for (int l = 0; l < L; ++l) {
      cvt_transpose_bf16<<<(nEl + 255) / 256, 256, 0, stream>>>(
          W1 + (size_t)l * nEl, wB1t + (size_t)l * nEl, 256, 512);
      cvt_transpose_bf16<<<(nEl + 255) / 256, 256, 0, stream>>>(
          W2 + (size_t)l * nEl, wB2t + (size_t)l * nEl, 512, 256);
    }
    cvt_transpose_bf16<<<(nEl + 255) / 256, 256, 0, stream>>>(Wp, wpBt, 512, 256);
  }

  // ---- node embeddings ----
  embed_sub<<<(NSUB * 64) / 256, 256, 0, stream>>>(x, sni, emb1, emb2, h, NSUB);
  hipMemsetAsync(out_cat, 0, (size_t)N * 512 * 4, stream);
  embed_origin<<<(N * 64) / 256, 256, 0, stream>>>(x, emb1, emb2, out_cat, N);

  // ---- GIN layers ----
  for (int l = 0; l < L; ++l) {
    hipMemsetAsync(agg, 0, (size_t)NSUB * 256 * 4, stream);
    scatter_edges<<<(ETOT * 64) / 256, 256, 0, stream>>>(
        h, src, dst, eattr, edge_e1 + (size_t)l * 6 * 256,
        edge_e2 + (size_t)l * 3 * 256, agg, ESUB, ETOT);

    dim3 g1g((NSUB + BLK_M - 1) / BLK_M, 512 / BLK_N);
    gemm_bf16_wmma<<<g1g, 256, 0, stream>>>(
        agg, wB1t + (size_t)l * 256 * 512, b1 + (size_t)l * 512,
        g1, NSUB, 512, 256, 1);

    dim3 g2g((NSUB + BLK_M - 1) / BLK_M, 256 / BLK_N);
    gemm_bf16_wmma<<<g2g, 256, 0, stream>>>(
        g1, wB2t + (size_t)l * 512 * 256, b2 + (size_t)l * 256,
        h, NSUB, 256, 512, 1);
  }

  // ---- pool subgraph nodes onto centers ----
  pool_nodes<<<(NSUB * 64) / 256, 256, 0, stream>>>(h, ind, out_cat, NSUB);

  // ---- batchnorm over concat features (elementwise), then projection GEMM ----
  hipMemsetAsync(bn1buf, 0, 1024 * 4, stream);
  col_stats<<<(N + 127) / 128, 256, 0, stream>>>(out_cat, N, 512, 128, sums1, sumsq1);
  bn_finalize<<<2, 256, 0, stream>>>(sums1, sumsq1, bn_cat_g, bn_cat_b,
                                     1.0f / N, 512, scale1, shift1);
  bn_apply<<<(N * 512) / 256, 256, 0, stream>>>(out_cat, scale1, shift1, N, 512);

  dim3 gpg((N + BLK_M - 1) / BLK_M, 256 / BLK_N);
  gemm_bf16_wmma<<<gpg, 256, 0, stream>>>(
      out_cat, wpBt, bp, (float*)d_out, N, 256, 512, 0);

  // ---- final encoder norm applied in place on d_out ----
  hipMemsetAsync(bn2buf, 0, 512 * 4, stream);
  col_stats<<<(N + 127) / 128, 256, 0, stream>>>((float*)d_out, N, 256, 128, sums2, sumsq2);
  bn_finalize<<<1, 256, 0, stream>>>(sums2, sumsq2, norm_g, norm_b,
                                     1.0f / N, 256, scale2, shift2);
  bn_apply<<<(N * 256) / 256, 256, 0, stream>>>((float*)d_out, scale2, shift2, N, 256);
}
